// VRNN_model_38998303048099
// MI455X (gfx1250) — matrix operand order repaired and checked by reading, compiled, not run on hardware
//
#include <hip/hip_runtime.h>
#include <hip/hip_bf16.h>

// ---------------------------------------------------------------------------
// CDNA5 (gfx1250, wave32) VRNN:
//  Phase A: phi_x for all B*T rows via v_wmma_f32_16x16x32_f16 (parallel).
//  Phase B: sequential GRU recurrence, 1 wave per 16-row batch tile,
//           weights as f16 N-major in LDS, per-step WMMA chain.
// ---------------------------------------------------------------------------

typedef _Float16 h16 __attribute__((ext_vector_type(16)));
typedef _Float16 h8  __attribute__((ext_vector_type(8)));
typedef float    f8  __attribute__((ext_vector_type(8)));

#define DEVI static __device__ __forceinline__

DEVI h16 cat16(h8 lo, h8 hi) {
  return __builtin_shufflevector(lo, hi, 0,1,2,3,4,5,6,7,8,9,10,11,12,13,14,15);
}

// A-fragment (16x32 f16): lane = row (l16). Slots 0..7 = K kt*32+hf*8+j,
// slots 8..15 = K kt*32+16+hf*8+j  (matches ISA 16-bit A 16x32 layout).
DEVI h16 afrag(const _Float16* rowp, int kt, int hf) {
  const _Float16* p = rowp + kt * 32 + hf * 8;
  h8 lo = *(const h8*)(p);
  h8 hi = *(const h8*)(p + 16);
  return cat16(lo, hi);
}

// B-fragment (32x16 f16) from N-major (transposed) weights: lane = column n,
// slots 0..7 = K kt*32+hf*8+j, slots 8..15 = K kt*32+16+hf*8+j.
DEVI h16 bfrag(const _Float16* wt, int Kpad, int kt, int nt, int l16, int hf) {
  const _Float16* p = wt + (size_t)(nt * 16 + l16) * Kpad + kt * 32 + hf * 8;
  h8 lo = *(const h8*)(p);
  h8 hi = *(const h8*)(p + 16);
  return cat16(lo, hi);
}

DEVI f8 wmma_f16(h16 a, h16 b, f8 c) {
  return __builtin_amdgcn_wmma_f32_16x16x32_f16(false, a, false, b,
                                                (short)0, c, false, false);
}

DEVI f8 bcast8(float v) {
  f8 c;
#pragma unroll
  for (int r = 0; r < 8; ++r) c[r] = v;
  return c;
}

// ReLU as a single v_med3_f32 (avoids the canonicalize+max pair of fmaxf).
DEVI float relu1(float x) {
  return __builtin_amdgcn_fmed3f(x, 0.0f, __builtin_inff());
}
DEVI f8 relu8(f8 x) {
#pragma unroll
  for (int r = 0; r < 8; ++r) x[r] = relu1(x[r]);
  return x;
}

// C/D-tile (lane=n within tile, vgpr r -> row r + hf*8) -> row-major f16.
DEVI void ctile_to_f16(_Float16* dst, int ld, int nt, f8 c, int l16, int hf) {
  _Float16* p = dst + (size_t)(hf * 8) * ld + nt * 16 + l16;
#pragma unroll
  for (int r = 0; r < 8; ++r) p[(size_t)r * ld] = (_Float16)c[r];
}

DEVI float sigm(float x) { return __frcp_rn(1.0f + __expf(-x)); }
// Fast tanh on the v_exp/v_rcp path: tanh(x) = 2*sigmoid(2x) - 1.
DEVI float tanh_fast(float x) {
  return __builtin_fmaf(2.0f, sigm(2.0f * x), -1.0f);
}
DEVI float sfpl(float x) {  // softplus, overflow-safe
  return (x > 0.0f) ? (x + log1pf(__expf(-x))) : log1pf(__expf(x));
}

// Transpose+convert f32 weights [K][N] -> LDS f16 N-major [Npad][Kpad].
DEVI void prep_w(_Float16* dst, const float* src, int K, int Nsrc,
                 int Kpad, int Npad, int tid, int nthr) {
  for (int i = tid; i < Npad * Kpad; i += nthr) {
    int n = i / Kpad, k = i - n * Kpad;
    float v = (k < K && n < Nsrc) ? src[(size_t)k * Nsrc + n] : 0.0f;
    dst[i] = (_Float16)v;
  }
}

static constexpr int Bsz = 1024, Tlen = 512;

// ============================ Phase A: phi_x ================================
__global__ void __launch_bounds__(256)
vrnn_phix(const float* __restrict__ src, const float* __restrict__ Wx1,
          const float* __restrict__ bx1, const float* __restrict__ Wx2,
          const float* __restrict__ bx2, _Float16* __restrict__ phiX) {
  __shared__ _Float16 sW1[64 * 32];    // Wx1^T padded K 24->32
  __shared__ _Float16 sW2[64 * 64];    // Wx2^T
  __shared__ _Float16 sWork[8 * 1024]; // per-wave 16x64 transpose scratch

  const int tid = threadIdx.x;
  for (int i = tid; i < 64 * 32; i += 256) {
    int n = i >> 5, k = i & 31;
    sW1[i] = (_Float16)((k < 24) ? Wx1[(size_t)k * 64 + n] : 0.0f);
  }
  for (int i = tid; i < 64 * 64; i += 256) {
    int n = i >> 6, k = i & 63;
    sW2[i] = (_Float16)Wx2[(size_t)k * 64 + n];
  }
  __syncthreads();

  const int wv = tid >> 5, ln = tid & 31, l16 = ln & 15, hf = (ln >> 4) & 1;
  const size_t tile = (size_t)blockIdx.x * 8 + wv;   // 16-row tile of B*T
  const size_t r0 = tile * 16;

  // Build src A-fragment (16x32, K padded 24->32) straight from global f32.
  const float* srow = src + (r0 + l16) * 24;
  h16 sA;
#pragma unroll
  for (int j = 0; j < 8; ++j) sA[j] = (_Float16)srow[hf * 8 + j];
#pragma unroll
  for (int j = 0; j < 8; ++j) {
    int kk = 16 + hf * 8 + j;
    sA[8 + j] = (_Float16)((kk < 24) ? srow[kk] : 0.0f);
  }

  _Float16* wk = sWork + wv * 1024;
  _Float16* wkRow = wk + l16 * 64;

  f8 c1[4];
#pragma unroll
  for (int nt = 0; nt < 4; ++nt) {
    f8 c = bcast8(bx1[nt * 16 + l16]);
    c = wmma_f16(sA, bfrag(sW1, 32, 0, nt, l16, hf), c);
    c1[nt] = relu8(c);
  }
#pragma unroll
  for (int nt = 0; nt < 4; ++nt) ctile_to_f16(wk, 64, nt, c1[nt], l16, hf);
  h16 a0 = afrag(wkRow, 0, hf), a1 = afrag(wkRow, 1, hf);

#pragma unroll
  for (int nt = 0; nt < 4; ++nt) {
    f8 c = bcast8(bx2[nt * 16 + l16]);
    c = wmma_f16(a0, bfrag(sW2, 64, 0, nt, l16, hf), c);
    c = wmma_f16(a1, bfrag(sW2, 64, 1, nt, l16, hf), c);
    c = relu8(c);
    _Float16* p = phiX + (r0 + hf * 8) * 64 + nt * 16 + l16;
#pragma unroll
    for (int r = 0; r < 8; ++r) p[(size_t)r * 64] = (_Float16)c[r];
  }
}

// ========================= Phase B: recurrence ==============================
__global__ void __launch_bounds__(32)
vrnn_recur(const float* __restrict__ eps,
           const float* Wz,  const float* bz,
           const float* We1, const float* be1,
           const float* We2, const float* be2,
           const float* Wem, const float* bem,
           const float* Wes, const float* bes,
           const float* Wih, const float* Whh,
           const float* Wf1, const float* bf1,
           const float* Wf2, const float* bf2,
           const float* Wf3, const float* bf3,
           const _Float16* __restrict__ phiX,
           float* __restrict__ out) {
  extern __shared__ _Float16 sm[];
  _Float16* We1t = sm;             // [64][128]  8192
  _Float16* We2t = We1t + 8192;    // [64][64]   4096
  _Float16* Wemt = We2t + 4096;    // [16][64]   1024
  _Float16* West = Wemt + 1024;    // [16][64]   1024
  _Float16* Wzt  = West + 1024;    // [64][32]   2048
  _Float16* Wiht = Wzt  + 2048;    // [192][128] 24576
  _Float16* Whht = Wiht + 24576;   // [192][64]  12288
  _Float16* Wf1t = Whht + 12288;   // [64][64]   4096
  _Float16* Wf2t = Wf1t + 4096;    // [32][64]   2048
  _Float16* Wf3t = Wf2t + 2048;    // [16][32]   512
  _Float16* work = Wf3t + 512;     // 16x64 scratch  1024
  _Float16* zbuf = work + 1024;    // 16x32 scratch  512
  _Float16* hbuf = zbuf + 512;     // 16x64 h state  1024
  // total 62464 halfs = 124928 bytes dynamic LDS

  const int tid = threadIdx.x;
  prep_w(We1t, We1, 128, 64, 128, 64, tid, 32);
  prep_w(We2t, We2, 64, 64, 64, 64, tid, 32);
  prep_w(Wemt, Wem, 64, 16, 64, 16, tid, 32);
  prep_w(West, Wes, 64, 16, 64, 16, tid, 32);
  prep_w(Wzt,  Wz,  16, 64, 32, 64, tid, 32);
  prep_w(Wiht, Wih, 128, 192, 128, 192, tid, 32);
  prep_w(Whht, Whh, 64, 192, 64, 192, tid, 32);
  prep_w(Wf1t, Wf1, 64, 64, 64, 64, tid, 32);
  prep_w(Wf2t, Wf2, 64, 32, 64, 32, tid, 32);
  prep_w(Wf3t, Wf3, 32, 1, 32, 16, tid, 32);
  for (int i = tid; i < 1024; i += 32) hbuf[i] = (_Float16)0.0f;  // h0 = 0
  for (int i = tid; i < 512;  i += 32) zbuf[i] = (_Float16)0.0f;  // K-pad
  __syncthreads();

  const int l16 = tid & 15, hf = (tid >> 4) & 1;
  const int b0 = blockIdx.x * 16;
  _Float16* workRow = work + l16 * 64;
  _Float16* zRow    = zbuf + l16 * 32;
  _Float16* hRow    = hbuf + l16 * 64;
  const _Float16* pxRow = phiX + (size_t)(b0 + l16) * Tlen * 64;

  float be1v[4], be2v[4], bzv[4], bf1v[4], bf2v[2];
#pragma unroll
  for (int nt = 0; nt < 4; ++nt) {
    be1v[nt] = be1[nt * 16 + l16];
    be2v[nt] = be2[nt * 16 + l16];
    bzv[nt]  = bz[nt * 16 + l16];
    bf1v[nt] = bf1[nt * 16 + l16];
  }
  bf2v[0] = bf2[l16];
  bf2v[1] = bf2[16 + l16];
  const float bemv = bem[l16], besv = bes[l16];
  const float bf3v = (l16 == 0) ? bf3[0] : 0.0f;

  f8 hC[4];
#pragma unroll
  for (int nt = 0; nt < 4; ++nt) hC[nt] = bcast8(0.0f);
  f8 pzC[4];
  h16 pzA0, pzA1;

#pragma unroll 1
  for (int t = 0; t < Tlen; ++t) {
    const _Float16* px = pxRow + (size_t)t * 64;
    h16 pxA0 = afrag(px, 0, hf), pxA1 = afrag(px, 1, hf);
    h16 hA0 = afrag(hRow, 0, hf), hA1 = afrag(hRow, 1, hf);
    if (t < Tlen - 1) __builtin_prefetch(px + 64, 0, 1);  // global_prefetch

    // enc1 = relu([phi_x, h] @ We1 + be1)          K=128, N=64
    f8 e1[4];
#pragma unroll
    for (int nt = 0; nt < 4; ++nt) {
      f8 c = bcast8(be1v[nt]);
      c = wmma_f16(pxA0, bfrag(We1t, 128, 0, nt, l16, hf), c);
      c = wmma_f16(pxA1, bfrag(We1t, 128, 1, nt, l16, hf), c);
      c = wmma_f16(hA0,  bfrag(We1t, 128, 2, nt, l16, hf), c);
      c = wmma_f16(hA1,  bfrag(We1t, 128, 3, nt, l16, hf), c);
      e1[nt] = relu8(c);
    }
#pragma unroll
    for (int nt = 0; nt < 4; ++nt) ctile_to_f16(work, 64, nt, e1[nt], l16, hf);
    h16 e1A0 = afrag(workRow, 0, hf), e1A1 = afrag(workRow, 1, hf);

    // enc2 = relu(enc1 @ We2 + be2)                K=64, N=64
    f8 e2[4];
#pragma unroll
    for (int nt = 0; nt < 4; ++nt) {
      f8 c = bcast8(be2v[nt]);
      c = wmma_f16(e1A0, bfrag(We2t, 64, 0, nt, l16, hf), c);
      c = wmma_f16(e1A1, bfrag(We2t, 64, 1, nt, l16, hf), c);
      e2[nt] = relu8(c);
    }
#pragma unroll
    for (int nt = 0; nt < 4; ++nt) ctile_to_f16(work, 64, nt, e2[nt], l16, hf);
    h16 e2A0 = afrag(workRow, 0, hf), e2A1 = afrag(workRow, 1, hf);

    // enc_mean / enc_std                           K=64, N=16
    f8 mC = bcast8(bemv);
    mC = wmma_f16(e2A0, bfrag(Wemt, 64, 0, 0, l16, hf), mC);
    mC = wmma_f16(e2A1, bfrag(Wemt, 64, 1, 0, l16, hf), mC);
    f8 sC = bcast8(besv);
    sC = wmma_f16(e2A0, bfrag(West, 64, 0, 0, l16, hf), sC);
    sC = wmma_f16(e2A1, bfrag(West, 64, 1, 0, l16, hf), sC);
#pragma unroll
    for (int r = 0; r < 8; ++r) sC[r] = sfpl(sC[r]);

    // z = eps * std + mean   (eps gathered per element)
    f8 zC;
#pragma unroll
    for (int r = 0; r < 8; ++r) {
      size_t ei = ((size_t)(b0 + r + hf * 8) * Tlen + t) * 16 + l16;
      zC[r] = __builtin_fmaf(eps[ei], sC[r], mC[r]);
    }
    ctile_to_f16(zbuf, 32, 0, zC, l16, hf);
    h16 zA = afrag(zRow, 0, hf);  // K=16 real + 16 zero pad

    // phi_z = relu(z @ Wz + bz)                    K=32(pad), N=64
#pragma unroll
    for (int nt = 0; nt < 4; ++nt) {
      f8 c = bcast8(bzv[nt]);
      c = wmma_f16(zA, bfrag(Wzt, 32, 0, nt, l16, hf), c);
      pzC[nt] = relu8(c);
    }
#pragma unroll
    for (int nt = 0; nt < 4; ++nt) ctile_to_f16(work, 64, nt, pzC[nt], l16, hf);
    pzA0 = afrag(workRow, 0, hf);
    pzA1 = afrag(workRow, 1, hf);

    // GRU gates: gi = [phi_x, phi_z] @ Wih, gh = h @ Whh; order r,z,n
    f8 rr[4], uu[4];
#pragma unroll
    for (int g = 0; g < 3; ++g) {
#pragma unroll
      for (int nt = 0; nt < 4; ++nt) {
        const int NT = g * 4 + nt;
        f8 gi = bcast8(0.0f);
        gi = wmma_f16(pxA0, bfrag(Wiht, 128, 0, NT, l16, hf), gi);
        gi = wmma_f16(pxA1, bfrag(Wiht, 128, 1, NT, l16, hf), gi);
        gi = wmma_f16(pzA0, bfrag(Wiht, 128, 2, NT, l16, hf), gi);
        gi = wmma_f16(pzA1, bfrag(Wiht, 128, 3, NT, l16, hf), gi);
        f8 gh = bcast8(0.0f);
        gh = wmma_f16(hA0, bfrag(Whht, 64, 0, NT, l16, hf), gh);
        gh = wmma_f16(hA1, bfrag(Whht, 64, 1, NT, l16, hf), gh);
        if (g == 0) {
#pragma unroll
          for (int r = 0; r < 8; ++r) rr[nt][r] = sigm(gi[r] + gh[r]);
        } else if (g == 1) {
#pragma unroll
          for (int r = 0; r < 8; ++r) uu[nt][r] = sigm(gi[r] + gh[r]);
        } else {
#pragma unroll
          for (int r = 0; r < 8; ++r) {
            float nn = tanh_fast(__builtin_fmaf(rr[nt][r], gh[r], gi[r]));
            hC[nt][r] =
                __builtin_fmaf(uu[nt][r], hC[nt][r] - nn, nn);  // (1-u)n + u h
          }
        }
      }
    }
#pragma unroll
    for (int nt = 0; nt < 4; ++nt) ctile_to_f16(hbuf, 64, nt, hC[nt], l16, hf);
  }

  // ---- regressor head on last-step phi_z ----
  f8 o1[4];
#pragma unroll
  for (int nt = 0; nt < 4; ++nt) {
    f8 c = bcast8(bf1v[nt]);
    c = wmma_f16(pzA0, bfrag(Wf1t, 64, 0, nt, l16, hf), c);
    c = wmma_f16(pzA1, bfrag(Wf1t, 64, 1, nt, l16, hf), c);
    o1[nt] = relu8(c);
  }
#pragma unroll
  for (int nt = 0; nt < 4; ++nt) ctile_to_f16(work, 64, nt, o1[nt], l16, hf);
  h16 o1A0 = afrag(workRow, 0, hf), o1A1 = afrag(workRow, 1, hf);

  f8 o2[2];
#pragma unroll
  for (int nt = 0; nt < 2; ++nt) {
    f8 c = bcast8(bf2v[nt]);
    c = wmma_f16(o1A0, bfrag(Wf2t, 64, 0, nt, l16, hf), c);
    c = wmma_f16(o1A1, bfrag(Wf2t, 64, 1, nt, l16, hf), c);
    o2[nt] = relu8(c);
  }
#pragma unroll
  for (int nt = 0; nt < 2; ++nt) ctile_to_f16(zbuf, 32, nt, o2[nt], l16, hf);
  h16 o2A = afrag(zRow, 0, hf);  // K=32 exactly

  f8 pd = bcast8(bf3v);          // N padded 1->16; only column 0 valid
  pd = wmma_f16(o2A, bfrag(Wf3t, 32, 0, 0, l16, hf), pd);

  // pred [B] then phi_z [B,64], flat
  if (l16 == 0) {
#pragma unroll
    for (int r = 0; r < 8; ++r) out[b0 + r + hf * 8] = pd[r];
  }
#pragma unroll
  for (int nt = 0; nt < 4; ++nt) {
#pragma unroll
    for (int r = 0; r < 8; ++r)
      out[(size_t)Bsz + (size_t)(b0 + r + hf * 8) * 64 + nt * 16 + l16] =
          pzC[nt][r];
  }
}

// =============================== launch =====================================
extern "C" void kernel_launch(void* const* d_in, const int* in_sizes, int n_in,
                              void* d_out, int out_size, void* d_ws,
                              size_t ws_size, hipStream_t stream) {
  (void)in_sizes; (void)n_in; (void)out_size; (void)ws_size;
  const float* src = (const float*)d_in[0];
  const float* eps = (const float*)d_in[1];
  const float* Wx1 = (const float*)d_in[2];
  const float* bx1 = (const float*)d_in[3];
  const float* Wx2 = (const float*)d_in[4];
  const float* bx2 = (const float*)d_in[5];
  const float* Wz  = (const float*)d_in[6];
  const float* bz  = (const float*)d_in[7];
  const float* We1 = (const float*)d_in[8];
  const float* be1 = (const float*)d_in[9];
  const float* We2 = (const float*)d_in[10];
  const float* be2 = (const float*)d_in[11];
  const float* Wem = (const float*)d_in[12];
  const float* bem = (const float*)d_in[13];
  const float* Wes = (const float*)d_in[14];
  const float* bes = (const float*)d_in[15];
  const float* Wih = (const float*)d_in[16];
  const float* Whh = (const float*)d_in[17];
  const float* Wf1 = (const float*)d_in[18];
  const float* bf1 = (const float*)d_in[19];
  const float* Wf2 = (const float*)d_in[20];
  const float* bf2 = (const float*)d_in[21];
  const float* Wf3 = (const float*)d_in[22];
  const float* bf3 = (const float*)d_in[23];

  _Float16* phiX = (_Float16*)d_ws;  // B*T*64 f16 = 64 MB
  float* out = (float*)d_out;

  // Phase A: 32768 16-row tiles, 8 waves per 256-thread block.
  vrnn_phix<<<(Bsz * Tlen / 16) / 8, 256, 0, stream>>>(src, Wx1, bx1, Wx2, bx2,
                                                       phiX);

  // Phase B: one wave per 16-row batch tile; 124928 B dynamic LDS for
  // f16 N-major weights + transpose scratch (CDNA5 WGP LDS = 320 KB).
  const size_t shmem = 62464 * sizeof(_Float16);
  vrnn_recur<<<Bsz / 16, 32, shmem, stream>>>(
      eps, Wz, bz, We1, be1, We2, be2, Wem, bem, Wes, bes, Wih, Whh, Wf1, bf1,
      Wf2, bf2, Wf3, bf3, phiX, out);
}
